// DynSMHALayer_3410204033646
// MI455X (gfx1250) — matrix operand order repaired and checked by reading, not compiled, not run on hardware
//
#include <hip/hip_runtime.h>
#include <stdint.h>

typedef __attribute__((ext_vector_type(16))) __bf16 v16bf;
typedef __attribute__((ext_vector_type(8)))  __bf16 v8bf;
typedef __attribute__((ext_vector_type(8)))  float  v8f;

#define Bb 4
#define Tt 2048
#define Cc 1024
#define Hh 128
#define Ee 16
#define Nn (Bb*Tt)

static __device__ __forceinline__ v16bf concat8(v8bf lo, v8bf hi){
    return __builtin_shufflevector(lo, hi, 0,1,2,3,4,5,6,7,8,9,10,11,12,13,14,15);
}
// A-fragment: elements 0..7 at p, elements 8..15 at p+16 (16B-aligned each)
static __device__ __forceinline__ v16bf load_afrag(const __bf16* p){
    return concat8(*(const v8bf*)p, *(const v8bf*)(p + 16));
}
// contiguous 16 bf16 (two 16B-aligned halves)
static __device__ __forceinline__ v16bf load_contig16(const __bf16* p){
    return concat8(*(const v8bf*)p, *(const v8bf*)(p + 8));
}
// CDNA5 async global->LDS copy of 16 bytes per lane
static __device__ __forceinline__ void async_copy16(unsigned ldsoff, const __bf16* g){
    unsigned long long ga = (unsigned long long)(size_t)g;
    asm volatile("global_load_async_to_lds_b128 %0, %1, off"
                 :: "v"(ldsoff), "v"(ga) : "memory");
}
static __device__ __forceinline__ void async_wait0(){
    asm volatile("s_wait_asynccnt 0x0" ::: "memory");
}
static __device__ __forceinline__ void async_wait4(){
    asm volatile("s_wait_asynccnt 0x4" ::: "memory");
}

// ---------------- Kernel 1: sim_matrix column inverse norms ----------------
__global__ void __launch_bounds__(256) colnorm_kernel(const float* __restrict__ sim,
                                                      float* __restrict__ sn_inv){
    int e = blockIdx.x;
    float acc = 0.f;
    for(int c = threadIdx.x; c < Cc; c += 256){
        float v = sim[(size_t)c*Ee + e];
        acc += v*v;
    }
    __shared__ float red[256];
    red[threadIdx.x] = acc;
    __syncthreads();
    for(int s = 128; s > 0; s >>= 1){
        if(threadIdx.x < s) red[threadIdx.x] += red[threadIdx.x + s];
        __syncthreads();
    }
    if(threadIdx.x == 0){
        float n = sqrtf(red[0]);
        sn_inv[e] = 1.0f / fmaxf(n, 1e-12f);
    }
}

// ---------------- Kernel 2: gating -> routing weights w[N,E] ----------------
__global__ void __launch_bounds__(256, 1) gating_kernel(const float* __restrict__ x,
                              const float* __restrict__ sim,
                              const float* __restrict__ gates, const float* __restrict__ sn_inv,
                              float* __restrict__ w){
    int token = blockIdx.x*8 + (threadIdx.x >> 5);
    int lane  = threadIdx.x & 31;
    const float* xr = x + (size_t)token*Cc;
    float dot[Ee];
    #pragma unroll
    for(int e = 0; e < Ee; e++) dot[e] = 0.f;
    float nrm = 0.f;
    for(int c = lane; c < Cc; c += 32){
        float xv = xr[c];
        nrm += xv*xv;
        const float* sr = sim + (size_t)c*Ee;
        #pragma unroll
        for(int e = 0; e < Ee; e++) dot[e] += xv*sr[e];
    }
    #pragma unroll
    for(int off = 16; off > 0; off >>= 1){
        nrm += __shfl_xor(nrm, off, 32);
        #pragma unroll
        for(int e = 0; e < Ee; e++) dot[e] += __shfl_xor(dot[e], off, 32);
    }
    if(lane == 0){
        float inx = 1.f / fmaxf(sqrtf(nrm), 1e-12f);
        float logits[Ee], gated[Ee];
        int active = 0;
        for(int e = 0; e < Ee; e++){
            float lg = dot[e]*inx*sn_inv[e] - 1.f/(1.f + __expf(-gates[e]));
            logits[e] = lg;
            gated[e]  = fmaxf(lg, 0.f);
            if(gated[e] > 0.f) active++;
        }
        bool mask[Ee];
        for(int e = 0; e < Ee; e++) mask[e] = gated[e] > 0.f;
        if(active == 0){
            int i0 = 0;
            for(int e = 1; e < Ee; e++) if(logits[e] > logits[i0]) i0 = e;
            int i1 = -1;
            for(int e = 0; e < Ee; e++){
                if(e == i0) continue;
                if(i1 < 0 || logits[e] > logits[i1]) i1 = e;
            }
            for(int e = 0; e < Ee; e++) mask[e] = false;
            mask[i0] = true; mask[i1] = true;
        }
        float mx = -1e30f;
        for(int e = 0; e < Ee; e++) if(mask[e]) mx = fmaxf(mx, gated[e]);
        float p[Ee], s = 0.f;
        for(int e = 0; e < Ee; e++){
            p[e] = mask[e] ? __expf(gated[e] - mx) : 0.f;
            s += p[e];
        }
        float* wr = w + (size_t)token*Ee;
        for(int e = 0; e < Ee; e++) wr[e] = p[e]/s;
    }
}

// ---------------- Kernel 3a: f32 -> bf16 (vectorized, 8 elems/thread) ----------------
__global__ void __launch_bounds__(256) cvt_bf16x8_kernel(const float* __restrict__ src,
                                                         unsigned short* __restrict__ dst, int n8){
    int i = blockIdx.x*256 + threadIdx.x;
    if(i >= n8) return;
    const float4* s4 = (const float4*)src;
    float4 a = s4[2*i], c = s4[2*i+1];
    v8bf o;
    o[0]=(__bf16)a.x; o[1]=(__bf16)a.y; o[2]=(__bf16)a.z; o[3]=(__bf16)a.w;
    o[4]=(__bf16)c.x; o[5]=(__bf16)c.y; o[6]=(__bf16)c.z; o[7]=(__bf16)c.w;
    *(v8bf*)((__bf16*)dst + (size_t)i*8) = o;
}

// ---------------- Kernel 3b: weight repack f32[E][K][Nd] -> fragment-major bf16 ----------------
__global__ void __launch_bounds__(256) repack_w_kernel(const float* __restrict__ src,
                                                       unsigned short* __restrict__ dst,
                                                       int K, int Nd){
    int tid  = blockIdx.x*256 + threadIdx.x;
    int lane = tid & 31;
    int frag = tid >> 5;
    int ntiles = Nd >> 4;
    int nt  = frag % ntiles;
    int ekb = frag / ntiles;
    int kbl = K >> 5;
    int kb  = ekb % kbl;
    int e   = ekb / kbl;
    int n     = nt*16 + (lane & 15);
    int kbase = kb*32 + (lane >> 4)*16;
    v8bf lo, hi;
    #pragma unroll
    for(int i = 0; i < 8; i++) lo[i] = (__bf16)src[((size_t)e*K + kbase + i)*Nd + n];
    #pragma unroll
    for(int i = 0; i < 8; i++) hi[i] = (__bf16)src[((size_t)e*K + kbase + 8 + i)*Nd + n];
    __bf16* d = (__bf16*)dst + (size_t)frag*512 + lane*16;
    *(v8bf*)d = lo;
    *(v8bf*)(d + 8) = hi;
}

// ---------------- Kernel 4: MoE projection  P[n,h] = sum_e w[n,e]*(x @ W_e) ----------------
// Two passes of 8 experts (64 accumulator VGPRs each) to keep everything register-resident;
// within a pass: k0 outer, experts inner, each LDS A-fragment feeds 8 back-to-back WMMAs.
#define XPITCH 1032   // 1024 + 8 bf16 pad (2064B rows -> bank stride 4)
__global__ void __launch_bounds__(256, 1)
moe_qkv_kernel(const unsigned short* __restrict__ Xbf_u,
               const unsigned short* __restrict__ Wbf_u,   // fragment-major
               const float* __restrict__ w,
               unsigned short* __restrict__ outbf_u,
               unsigned short* __restrict__ outT_u){       // optional [B][H][T]
    const __bf16* Xbf = (const __bf16*)Xbf_u;
    const __bf16* Wbf = (const __bf16*)Wbf_u;
    __bf16* outbf = (__bf16*)outbf_u;
    __shared__ __bf16 xtile[16*XPITCH];
    int mt   = blockIdx.x;
    int wv   = threadIdx.x >> 5;
    int lane = threadIdx.x & 31;
    int half = lane >> 4, l16 = lane & 15;
    int tokenBase = mt*16;
    // async stage X tile (16 x 1024 bf16) into LDS: 2048 16B-chunks, 8 per thread
    unsigned xt_off = (unsigned)(size_t)&xtile[0];
    #pragma unroll
    for(int j = 0; j < 8; j++){
        int c   = threadIdx.x + 256*j;
        int row = c >> 7, col = c & 127;
        async_copy16(xt_off + (unsigned)(row*XPITCH*2 + col*16),
                     Xbf + (size_t)(tokenBase + row)*Cc + col*8);
    }
    async_wait0();
    __syncthreads();

    v8f acc = (v8f){0,0,0,0,0,0,0,0};
    #pragma unroll 1
    for(int pass = 0; pass < 2; pass++){
        v8f eacc[8];
        #pragma unroll
        for(int e = 0; e < 8; e++) eacc[e] = (v8f){0,0,0,0,0,0,0,0};
        for(int k0 = 0; k0 < Cc; k0 += 32){
            v16bf a = load_contig16(&xtile[(size_t)l16*XPITCH + k0 + half*16]);
            const __bf16* bc = Wbf
                + (((size_t)(pass*8)*(Cc/32) + (k0 >> 5))*(Hh/16) + wv)*512 + lane*16;
            #pragma unroll
            for(int e = 0; e < 8; e++){
                v16bf bfr = load_contig16(bc + (size_t)e*(Cc/32)*(Hh/16)*512);
                eacc[e] = __builtin_amdgcn_wmma_f32_16x16x32_bf16(false, a, false, bfr,
                                                                  (short)0, eacc[e], false, false);
            }
        }
        #pragma unroll
        for(int r = 0; r < 8; r++){
            const float* wr = w + (size_t)(tokenBase + r + 8*half)*Ee + pass*8;
            #pragma unroll
            for(int e = 0; e < 8; e++) acc[r] += wr[e]*eacc[e][r];
        }
    }
    int h = wv*16 + l16;
    #pragma unroll
    for(int r = 0; r < 8; r++){
        int token = tokenBase + r + 8*half;
        outbf[(size_t)token*Hh + h] = (__bf16)acc[r];
    }
    if(outT_u){
        __bf16* outT = (__bf16*)outT_u;
        #pragma unroll
        for(int r = 0; r < 8; r++){
            int token = tokenBase + r + 8*half;
            int bidx = token >> 11;
            int tpos = token & (Tt - 1);
            outT[((size_t)bidx*Hh + h)*Tt + tpos] = (__bf16)acc[r];
        }
    }
}

// ---------------- Kernel 5: flash attention (single head, H=128) ----------------
#define KPITCH 136    // 128 + 8 pad
#define VPITCH 40     // 32 + 8 pad
__global__ void __launch_bounds__(256, 1)
attn_kernel(const unsigned short* __restrict__ qbf_u,
            const unsigned short* __restrict__ kbf_u,
            const unsigned short* __restrict__ vtbf_u,   // [B][H][T]
            unsigned short* __restrict__ obf_u){
    const __bf16* qbf = (const __bf16*)qbf_u;
    const __bf16* kbf = (const __bf16*)kbf_u;
    const __bf16* vt  = (const __bf16*)vtbf_u;
    __bf16* obf = (__bf16*)obf_u;
    __shared__ __bf16 ktile[2][32*KPITCH];
    __shared__ __bf16 vtile[2][Hh*VPITCH];
    __shared__ __bf16 pstage[8][16*VPITCH];
    int wv   = threadIdx.x >> 5;
    int lane = threadIdx.x & 31;
    int half = lane >> 4, l16 = lane & 15;
    int bidx = blockIdx.y;
    int qtile = blockIdx.x*8 + wv;
    unsigned kt_off[2] = { (unsigned)(size_t)&ktile[0][0], (unsigned)(size_t)&ktile[1][0] };
    unsigned vt_off[2] = { (unsigned)(size_t)&vtile[0][0], (unsigned)(size_t)&vtile[1][0] };

    auto issue_stage = [&](int kc, int buf){
        const __bf16* K0  = kbf + ((size_t)bidx*Tt + kc)*Hh;
        const __bf16* Vt0 = vt  + ((size_t)bidx*Hh)*Tt + kc;
        #pragma unroll
        for(int j = 0; j < 2; j++){
            int c = threadIdx.x + 256*j;
            int row = c >> 4, col = c & 15;
            async_copy16(kt_off[buf] + (unsigned)(row*KPITCH*2 + col*16),
                         K0 + (size_t)row*Hh + col*8);
        }
        #pragma unroll
        for(int j = 0; j < 2; j++){
            int c = threadIdx.x + 256*j;
            int hh = c >> 2, col = c & 3;
            async_copy16(vt_off[buf] + (unsigned)(hh*VPITCH*2 + col*16),
                         Vt0 + (size_t)hh*Tt + col*8);
        }
    };

    const __bf16* Q = qbf + ((size_t)bidx*Tt + qtile*16)*Hh;
    v16bf qa[4];
    #pragma unroll
    for(int s = 0; s < 4; s++)
        qa[s] = load_contig16(Q + (size_t)l16*Hh + s*32 + half*16);

    float rmax[8], rsum[8];
    v8f oacc[8];
    #pragma unroll
    for(int r = 0; r < 8; r++){ rmax[r] = -1e30f; rsum[r] = 0.f; }
    #pragma unroll
    for(int t = 0; t < 8; t++) oacc[t] = (v8f){0,0,0,0,0,0,0,0};
    const float scale = 0.08838834764831845f;   // 1/sqrt(128)

    const int niter = Tt/32;
    issue_stage(0, 0);
    for(int it = 0; it < niter; it++){
        int cur = it & 1;
        if(it + 1 < niter){
            issue_stage((it + 1)*32, cur ^ 1);
            async_wait4();            // previous stage's 4 ops complete (in-order)
        } else {
            async_wait0();
        }
        __syncthreads();

        const __bf16* kt  = &ktile[cur][0];
        const __bf16* vtl = &vtile[cur][0];

        v8f s0 = {0,0,0,0,0,0,0,0}, s1 = {0,0,0,0,0,0,0,0};
        #pragma unroll
        for(int s = 0; s < 4; s++){
            v16bf b0 = load_contig16(kt + (size_t)l16*KPITCH + s*32 + half*16);
            v16bf b1 = load_contig16(kt + (size_t)(16 + l16)*KPITCH + s*32 + half*16);
            s0 = __builtin_amdgcn_wmma_f32_16x16x32_bf16(false, qa[s], false, b0,
                                                         (short)0, s0, false, false);
            s1 = __builtin_amdgcn_wmma_f32_16x16x32_bf16(false, qa[s], false, b1,
                                                         (short)0, s1, false, false);
        }
        float m0[8];
        #pragma unroll
        for(int r = 0; r < 8; r++) m0[r] = fmaxf(s0[r], s1[r])*scale;
        #pragma unroll
        for(int off = 1; off < 16; off <<= 1){
            #pragma unroll
            for(int r = 0; r < 8; r++) m0[r] = fmaxf(m0[r], __shfl_xor(m0[r], off, 32));
        }
        float alpha[8], p0[8], p1[8], rs[8];
        #pragma unroll
        for(int r = 0; r < 8; r++){
            float nm = fmaxf(rmax[r], m0[r]);
            alpha[r] = __expf(rmax[r] - nm);
            rmax[r]  = nm;
            p0[r] = __expf(s0[r]*scale - nm);
            p1[r] = __expf(s1[r]*scale - nm);
            rs[r] = p0[r] + p1[r];
        }
        #pragma unroll
        for(int off = 1; off < 16; off <<= 1){
            #pragma unroll
            for(int r = 0; r < 8; r++) rs[r] += __shfl_xor(rs[r], off, 32);
        }
        #pragma unroll
        for(int r = 0; r < 8; r++) rsum[r] = rsum[r]*alpha[r] + rs[r];
        #pragma unroll
        for(int t = 0; t < 8; t++){
            #pragma unroll
            for(int r = 0; r < 8; r++) oacc[t][r] *= alpha[r];
        }
        __bf16* ps = &pstage[wv][0];
        #pragma unroll
        for(int r = 0; r < 8; r++){
            int m = r + 8*half;
            ps[m*VPITCH + l16]      = (__bf16)p0[r];
            ps[m*VPITCH + 16 + l16] = (__bf16)p1[r];
        }
        v16bf pa = load_contig16(ps + l16*VPITCH + half*16);
        #pragma unroll
        for(int t = 0; t < 8; t++){
            v16bf vb = load_contig16(vtl + (size_t)(t*16 + l16)*VPITCH + half*16);
            oacc[t] = __builtin_amdgcn_wmma_f32_16x16x32_bf16(false, pa, false, vb,
                                                              (short)0, oacc[t], false, false);
        }
        __syncthreads();   // protect buffers before next stage issue
    }
    #pragma unroll
    for(int t = 0; t < 8; t++){
        #pragma unroll
        for(int r = 0; r < 8; r++)
            obf[((size_t)bidx*Tt + qtile*16 + r + 8*half)*Hh + t*16 + l16] =
                (__bf16)(oacc[t][r]/rsum[r]);
    }
}

// ---------------- Kernel 6: MoE output combine  out[n,c] = sum_e w[n,e]*(o @ Wo_e) ----------------
#define OPITCH 136
__global__ void __launch_bounds__(256, 1)
moe_out_kernel(const unsigned short* __restrict__ obf_u,
               const unsigned short* __restrict__ Wo_u,   // fragment-major
               const float* __restrict__ w,
               float* __restrict__ out){
    const __bf16* obf = (const __bf16*)obf_u;
    const __bf16* Wo  = (const __bf16*)Wo_u;
    __shared__ __bf16 otile[16*OPITCH];
    int mt   = blockIdx.x;
    int wv   = threadIdx.x >> 5;
    int lane = threadIdx.x & 31;
    int half = lane >> 4, l16 = lane & 15;
    int ct   = blockIdx.y*8 + wv;       // 0..63
    int tokenBase = mt*16;
    unsigned ot_off = (unsigned)(size_t)&otile[0];
    {
        int c = threadIdx.x;
        int row = c >> 4, col = c & 15;
        async_copy16(ot_off + (unsigned)(row*OPITCH*2 + col*16),
                     obf + (size_t)(tokenBase + row)*Hh + col*8);
    }
    async_wait0();
    __syncthreads();

    v16bf oa[4];
    #pragma unroll
    for(int s = 0; s < 4; s++)
        oa[s] = load_contig16(&otile[(size_t)l16*OPITCH + s*32 + half*16]);

    v8f acc = (v8f){0,0,0,0,0,0,0,0};
    #pragma unroll 1
    for(int pass = 0; pass < 2; pass++){
        v8f eacc[8];
        #pragma unroll
        for(int e = 0; e < 8; e++) eacc[e] = (v8f){0,0,0,0,0,0,0,0};
        #pragma unroll
        for(int s = 0; s < 4; s++){
            const __bf16* bc = Wo + ((size_t)s*(Cc/16) + ct)*512 + lane*16
                               + (size_t)(pass*8)*(Hh/32)*(Cc/16)*512;
            #pragma unroll
            for(int e = 0; e < 8; e++){
                v16bf bfr = load_contig16(bc + (size_t)e*(Hh/32)*(Cc/16)*512);
                eacc[e] = __builtin_amdgcn_wmma_f32_16x16x32_bf16(false, oa[s], false, bfr,
                                                                  (short)0, eacc[e], false, false);
            }
        }
        #pragma unroll
        for(int r = 0; r < 8; r++){
            const float* wr = w + (size_t)(tokenBase + r + 8*half)*Ee + pass*8;
            #pragma unroll
            for(int e = 0; e < 8; e++) acc[r] += wr[e]*eacc[e][r];
        }
    }
    #pragma unroll
    for(int r = 0; r < 8; r++)
        out[(size_t)(tokenBase + r + 8*half)*Cc + ct*16 + l16] = acc[r];
}

// ---------------- workspace layout ----------------
static constexpr size_t OFF_W  = 256;
static constexpr size_t OFF_X  = OFF_W  + (size_t)Nn*Ee*4;
static constexpr size_t OFF_WQ = OFF_X  + (size_t)Nn*Cc*2;
static constexpr size_t OFF_WK = OFF_WQ + (size_t)Ee*Cc*Hh*2;
static constexpr size_t OFF_WV = OFF_WK + (size_t)Ee*Cc*Hh*2;
static constexpr size_t OFF_WO = OFF_WV + (size_t)Ee*Cc*Hh*2;
static constexpr size_t OFF_QB = OFF_WO + (size_t)Ee*Hh*Cc*2;
static constexpr size_t OFF_KB = OFF_QB + (size_t)Nn*Hh*2;
static constexpr size_t OFF_VB = OFF_KB + (size_t)Nn*Hh*2;
static constexpr size_t OFF_VT = OFF_VB + (size_t)Nn*Hh*2;
static constexpr size_t OFF_OB = OFF_VT + (size_t)Nn*Hh*2;

extern "C" void kernel_launch(void* const* d_in, const int* in_sizes, int n_in,
                              void* d_out, int out_size, void* d_ws, size_t ws_size,
                              hipStream_t stream){
    const float* x     = (const float*)d_in[0];
    const float* sim   = (const float*)d_in[1];
    const float* gates = (const float*)d_in[2];
    const float* qp    = (const float*)d_in[3];
    const float* kp    = (const float*)d_in[4];
    const float* vp    = (const float*)d_in[5];
    const float* op    = (const float*)d_in[6];
    float* out = (float*)d_out;
    char* ws = (char*)d_ws;

    float* sn_inv = (float*)ws;
    float* w      = (float*)(ws + OFF_W);
    unsigned short* Xbf = (unsigned short*)(ws + OFF_X);
    unsigned short* Wq  = (unsigned short*)(ws + OFF_WQ);
    unsigned short* Wk  = (unsigned short*)(ws + OFF_WK);
    unsigned short* Wv  = (unsigned short*)(ws + OFF_WV);
    unsigned short* Wo  = (unsigned short*)(ws + OFF_WO);
    unsigned short* qb  = (unsigned short*)(ws + OFF_QB);
    unsigned short* kb  = (unsigned short*)(ws + OFF_KB);
    unsigned short* vb  = (unsigned short*)(ws + OFF_VB);
    unsigned short* vt  = (unsigned short*)(ws + OFF_VT);
    unsigned short* ob  = (unsigned short*)(ws + OFF_OB);

    colnorm_kernel<<<Ee, 256, 0, stream>>>(sim, sn_inv);
    gating_kernel<<<Nn/8, 256, 0, stream>>>(x, sim, gates, sn_inv, w);
    {
        int n8 = Nn*Cc/8;
        cvt_bf16x8_kernel<<<(n8 + 255)/256, 256, 0, stream>>>(x, Xbf, n8);
        int nthr = Ee*Cc*Hh/16;
        repack_w_kernel<<<nthr/256, 256, 0, stream>>>(qp, Wq, Cc, Hh);
        repack_w_kernel<<<nthr/256, 256, 0, stream>>>(kp, Wk, Cc, Hh);
        repack_w_kernel<<<nthr/256, 256, 0, stream>>>(vp, Wv, Cc, Hh);
        repack_w_kernel<<<nthr/256, 256, 0, stream>>>(op, Wo, Hh, Cc);
    }
    moe_qkv_kernel<<<Nn/16, 256, 0, stream>>>(Xbf, Wq, w, qb, (unsigned short*)nullptr);
    moe_qkv_kernel<<<Nn/16, 256, 0, stream>>>(Xbf, Wk, w, kb, (unsigned short*)nullptr);
    moe_qkv_kernel<<<Nn/16, 256, 0, stream>>>(Xbf, Wv, w, vb, vt);
    attn_kernel<<<dim3(Tt/128, Bb), 256, 0, stream>>>(qb, kb, vt, ob);
    moe_out_kernel<<<dim3(Nn/16, 8), 256, 0, stream>>>(ob, Wo, w, out);
}
